// LLEGraphNet_57123065037607
// MI455X (gfx1250) — compile-verified, matrix operand order checked by reading, not run on hardware
//
#include <hip/hip_runtime.h>
#include <math.h>

// ---------------------------------------------------------------------------
// CDNA5 (gfx1250) wave32 WMMA GNN forward.
// All GEMMs use v_wmma_f32_16x16x32_bf16 (f32 accumulate).
// ---------------------------------------------------------------------------

typedef __attribute__((ext_vector_type(16))) __bf16 v16bf;
typedef __attribute__((ext_vector_type(8)))  float  v8f;

union Frag {
  unsigned int u[8];
  v16bf v;
};

__device__ __forceinline__ unsigned short bf16s(float x){
  unsigned u = __builtin_bit_cast(unsigned, x);
  u = (u + 0x7FFFu + ((u >> 16) & 1u)) >> 16;   // round-to-nearest-even
  return (unsigned short)u;
}
__device__ __forceinline__ unsigned pk_bf16(float a, float b){
  return (unsigned)bf16s(a) | ((unsigned)bf16s(b) << 16);
}
__device__ __forceinline__ float bf16lo(unsigned u){ return __builtin_bit_cast(float, u << 16); }
__device__ __forceinline__ float bf16hi(unsigned u){ return __builtin_bit_cast(float, u & 0xFFFF0000u); }

__device__ __forceinline__ float gelu_f(float x){
  return 0.5f * x * (1.0f + erff(x * 0.70710678118654752f));
}
__device__ __forceinline__ v8f zero8(){ v8f z = {0.f,0.f,0.f,0.f,0.f,0.f,0.f,0.f}; return z; }

__device__ __forceinline__ v8f wmma_bf16(Frag a, Frag b, v8f c){
  return __builtin_amdgcn_wmma_f32_16x16x32_bf16(false, a.v, false, b.v, (short)0, c, false, false);
}

// B fragment: packed weights, ((kt*ntiles+nt)*32 + lane)*8 consecutive u32.
__device__ __forceinline__ Frag load_bfrag(const unsigned* __restrict__ Wp,
                                           int ntiles, int kt, int nt, int lane){
  Frag f;
  const unsigned* p = Wp + ((size_t)((kt*ntiles + nt)*32 + lane) << 3);
  #pragma unroll
  for (int j = 0; j < 8; ++j) f.u[j] = p[j];
  return f;
}

// A fragment from LDS rows of bf16 pairs (u32).  ISA 16-bit A 16x32 layout:
// lane L: m = L&15, half = L>>4;  VGPR j holds K = (j>>2)*16 + half*8 + (j&3)*2, +1
__device__ __forceinline__ Frag lds_afrag_u32(const unsigned* s, int rowstride,
                                              int k0, int lane){
  Frag f; const int m = lane & 15, h = lane >> 4;
  #pragma unroll
  for (int j = 0; j < 8; ++j){
    int k = k0 + ((j >> 2) << 4) + (h << 3) + ((j & 3) << 1);
    f.u[j] = s[m*rowstride + (k >> 1)];
  }
  return f;
}

__device__ __forceinline__ Frag lds_afrag_u16(const unsigned short* s, int rowstride,
                                              int k0, int lane){
  Frag f; const int m = lane & 15, h = lane >> 4;
  #pragma unroll
  for (int j = 0; j < 8; ++j){
    int k = k0 + ((j >> 2) << 4) + (h << 3) + ((j & 3) << 1);
    f.u[j] = *(const unsigned*)(&s[m*rowstride + k]);   // k even -> 4B aligned
  }
  return f;
}

__device__ __forceinline__ void atomAddF(float* p, float v){
  __hip_atomic_fetch_add(p, v, __ATOMIC_RELAXED, __HIP_MEMORY_SCOPE_AGENT);
}

// ---------------------------------------------------------------------------
// Weight repack: f32 (din x dout) row-major -> bf16 B-fragment layout, K padded
// to ktiles*32 with zeros.
// ---------------------------------------------------------------------------
__global__ void pack_weight_kernel(const float* __restrict__ W, unsigned* __restrict__ out,
                                   int din, int dout, int ktiles){
  int ntiles = dout >> 4;
  int total = ktiles * ntiles * 256;
  for (int idx = blockIdx.x*blockDim.x + threadIdx.x; idx < total; idx += gridDim.x*blockDim.x){
    int j    = idx & 7;
    int lane = (idx >> 3) & 31;
    int t    = idx >> 8;                 // kt*ntiles + nt
    int nt   = t % ntiles;
    int kt   = t / ntiles;
    int n = nt*16 + (lane & 15);
    int k = kt*32 + ((j >> 2) << 4) + ((lane >> 4) << 3) + ((j & 3) << 1);
    float a = (k     < din) ? W[(size_t)k*dout + n]     : 0.f;
    float b = (k + 1 < din) ? W[(size_t)(k+1)*dout + n] : 0.f;
    out[idx] = pk_bf16(a, b);
  }
}

__global__ void f32_to_bf16pk(const float* __restrict__ in, unsigned* __restrict__ out, int npairs){
  int i = blockIdx.x*blockDim.x + threadIdx.x;
  if (i < npairs) out[i] = pk_bf16(in[2*i], in[2*i + 1]);
}

// ---------------------------------------------------------------------------
// out = gelu(x @ W + b) : x (M x K) f32, out (M x 128) f32.  One wave / 16 rows.
// ---------------------------------------------------------------------------
__global__ __launch_bounds__(32) void in_proj_kernel(const float* __restrict__ x, int K,
                                                     const unsigned* __restrict__ Wp,
                                                     const float* __restrict__ bias,
                                                     float* __restrict__ out){
  __shared__ unsigned s_a[16*32];
  const int lane = threadIdx.x;
  const long base = (long)blockIdx.x * 16;
  const int row = lane & 15, half = lane >> 4;
  {
    const float* xr = x + (base + row) * K;
    const int per = K >> 2;                 // u32 per half-row
    for (int i = half*per; i < (half + 1)*per; ++i)
      s_a[row*32 + i] = pk_bf16(xr[2*i], xr[2*i + 1]);
  }
  __syncthreads();
  const int KT = K >> 5;
  for (int nt = 0; nt < 8; ++nt){
    v8f c = zero8();
    for (int kt = 0; kt < KT; ++kt){
      Frag a = lds_afrag_u32(s_a, 32, kt*32, lane);
      Frag b = load_bfrag(Wp, 8, kt, nt, lane);
      c = wmma_bf16(a, b, c);
    }
    int n = nt*16 + row; float bv = bias[n];
    #pragma unroll
    for (int r = 0; r < 8; ++r)
      out[(base + r + 8*half)*128 + n] = gelu_f(c[r] + bv);
  }
}

// ---------------------------------------------------------------------------
// Edge kernel: one wave per 16-edge tile.  Gate + edge-MLP + LN(e) + message
// MLP + atomic scatter-add into agg[dst].  82 WMMAs per tile.
// ---------------------------------------------------------------------------
__global__ __launch_bounds__(32) void edge_kernel(
    const unsigned* __restrict__ hbf,          // N x 64 u32 (bf16 pairs of H=128)
    float* __restrict__ e,                     // E x 16 (in/out, in place)
    const int* __restrict__ src, const int* __restrict__ dst,
    const unsigned* __restrict__ We1, const float* __restrict__ be1,
    const unsigned* __restrict__ We2, const float* __restrict__ be2,
    const float* __restrict__ en_b, const float* __restrict__ en_g,
    const float* __restrict__ Wg,  const float* __restrict__ bg,
    const unsigned* __restrict__ Wm1, const float* __restrict__ bm1,
    const unsigned* __restrict__ Wm2, const float* __restrict__ bm2,
    float* __restrict__ agg){
  __shared__ unsigned       s_hs[16*64];
  __shared__ unsigned       s_hd[16*64];
  __shared__ float          s_ef[16*16];
  __shared__ unsigned       s_ebf[16*8];
  __shared__ float          s_t16[16*16];
  __shared__ unsigned short s_m1h[16*128];
  __shared__ int            s_idx[32];
  __shared__ float          s_gate[16];

  const int lane = threadIdx.x;
  const long ebase = (long)blockIdx.x * 16;
  const int row = lane & 15, half = lane >> 4;

  if (lane < 16){
    s_idx[lane]      = src[ebase + lane];
    s_idx[16 + lane] = dst[ebase + lane];
  }
  __syncthreads();

  { // gather h[src], h[dst] tiles (bf16), 32 u32 per lane per matrix
    long rs = (long)s_idx[row]*64 + half*32;
    long rd = (long)s_idx[16 + row]*64 + half*32;
    #pragma unroll
    for (int i = 0; i < 8; ++i){
      *(uint4*)&s_hs[row*64 + half*32 + i*4] = *(const uint4*)(hbf + rs + i*4);
      *(uint4*)&s_hd[row*64 + half*32 + i*4] = *(const uint4*)(hbf + rd + i*4);
    }
  }
  { // stage e tile (f32)
    int r = lane >> 1, c0 = (lane & 1)*8;
    #pragma unroll
    for (int i = 0; i < 8; ++i) s_ef[r*16 + c0 + i] = e[(ebase + r)*16 + c0 + i];
  }
  __syncthreads();

  // gate = sigmoid(z . Wg + bg),  z = [hs | hd | e]
  if (lane < 16){
    float s = bg[0];
    #pragma unroll 8
    for (int i = 0; i < 64; ++i){
      unsigned u = s_hs[lane*64 + i];
      s += bf16lo(u)*Wg[2*i] + bf16hi(u)*Wg[2*i + 1];
    }
    #pragma unroll 8
    for (int i = 0; i < 64; ++i){
      unsigned u = s_hd[lane*64 + i];
      s += bf16lo(u)*Wg[128 + 2*i] + bf16hi(u)*Wg[129 + 2*i];
    }
    #pragma unroll
    for (int i = 0; i < 16; ++i) s += s_ef[lane*16 + i]*Wg[256 + i];
    s_gate[lane] = 1.f / (1.f + expf(-s));
  }

  // ---- e1: (16 x 272) @ (272 x 16), K padded to 288 (9 ktiles) ----
  v8f d1 = zero8();
  #pragma unroll
  for (int kt = 0; kt < 4; ++kt){
    Frag a = lds_afrag_u32(s_hs, 64, kt*32, lane);
    d1 = wmma_bf16(a, load_bfrag(We1, 1, kt, 0, lane), d1);
  }
  #pragma unroll
  for (int kt = 0; kt < 4; ++kt){
    Frag a = lds_afrag_u32(s_hd, 64, kt*32, lane);
    d1 = wmma_bf16(a, load_bfrag(We1, 1, 4 + kt, 0, lane), d1);
  }
  {
    Frag a;
    #pragma unroll
    for (int j = 0; j < 8; ++j){
      int k = ((j >> 2) << 4) + (half << 3) + ((j & 3) << 1);
      a.u[j] = (k < 16) ? pk_bf16(s_ef[row*16 + k], s_ef[row*16 + k + 1]) : 0u;
    }
    d1 = wmma_bf16(a, load_bfrag(We1, 1, 8, 0, lane), d1);
  }
  { // gelu(e1 + b) -> s_t16
    float bv = be1[row];
    #pragma unroll
    for (int r = 0; r < 8; ++r) s_t16[(r + 8*half)*16 + row] = gelu_f(d1[r] + bv);
  }
  __syncthreads();

  // ---- e2: (16 x 16) @ (16 x 16), K padded to 32 ----
  v8f d2 = zero8();
  {
    Frag a;
    #pragma unroll
    for (int j = 0; j < 8; ++j){
      int k = ((j >> 2) << 4) + (half << 3) + ((j & 3) << 1);
      a.u[j] = (k < 16) ? pk_bf16(s_t16[row*16 + k], s_t16[row*16 + k + 1]) : 0u;
    }
    d2 = wmma_bf16(a, load_bfrag(We2, 1, 0, 0, lane), d2);
  }
  __syncthreads();
  { // e_tmp = e + EDGE_SCALE * delta * gate  -> s_t16 (reuse)
    float bv = be2[row];
    #pragma unroll
    for (int r = 0; r < 8; ++r){
      int m = r + 8*half;
      s_t16[m*16 + row] = s_ef[m*16 + row] + 0.1f*s_gate[m]*(d2[r] + bv);
    }
  }
  __syncthreads();

  // ---- LayerNorm(e) per edge; write back + bf16 stage for msg1 ----
  if (lane < 16){
    float mu = 0.f;
    #pragma unroll
    for (int c = 0; c < 16; ++c) mu += s_t16[lane*16 + c];
    mu *= (1.f/16.f);
    float var = 0.f;
    #pragma unroll
    for (int c = 0; c < 16; ++c){ float d = s_t16[lane*16 + c] - mu; var += d*d; }
    var *= (1.f/16.f);
    float inv = rsqrtf(var + 1e-5f);
    float tmp[16];
    #pragma unroll
    for (int c = 0; c < 16; ++c) tmp[c] = (s_t16[lane*16 + c] - mu)*inv*en_g[c] + en_b[c];
    #pragma unroll
    for (int c = 0; c < 16; ++c) e[(ebase + lane)*16 + c] = tmp[c];
    #pragma unroll
    for (int c = 0; c < 8; ++c) s_ebf[lane*8 + c] = pk_bf16(tmp[2*c], tmp[2*c + 1]);
  }
  __syncthreads();

  // ---- msg1: (16 x 160) @ (160 x 128) ----
  #pragma unroll 1
  for (int nt = 0; nt < 8; ++nt){
    v8f c = zero8();
    #pragma unroll
    for (int kt = 0; kt < 4; ++kt){
      Frag a = lds_afrag_u32(s_hs, 64, kt*32, lane);
      c = wmma_bf16(a, load_bfrag(Wm1, 8, kt, nt, lane), c);
    }
    {
      Frag a;
      #pragma unroll
      for (int j = 0; j < 8; ++j){
        int k = ((j >> 2) << 4) + (half << 3) + ((j & 3) << 1);
        a.u[j] = (k < 16) ? s_ebf[row*8 + (k >> 1)] : 0u;
      }
      c = wmma_bf16(a, load_bfrag(Wm1, 8, 4, nt, lane), c);
    }
    int n = nt*16 + row; float bv = bm1[n];
    #pragma unroll
    for (int r = 0; r < 8; ++r) s_m1h[(r + 8*half)*128 + n] = bf16s(gelu_f(c[r] + bv));
  }
  __syncthreads();

  // ---- msg2: (16 x 128) @ (128 x 128) + scatter-add into agg[dst] ----
  #pragma unroll 1
  for (int nt = 0; nt < 8; ++nt){
    v8f c = zero8();
    #pragma unroll
    for (int kt = 0; kt < 4; ++kt){
      Frag a = lds_afrag_u16(s_m1h, 128, kt*32, lane);
      c = wmma_bf16(a, load_bfrag(Wm2, 8, kt, nt, lane), c);
    }
    int n = nt*16 + row; float bv = bm2[n];
    #pragma unroll
    for (int r = 0; r < 8; ++r){
      int m = r + 8*half;
      atomAddF(&agg[(long)s_idx[16 + m]*128 + n], c[r] + bv);
    }
  }
}

// ---------------------------------------------------------------------------
// Node update: h = LN(h + upd2(gelu(upd1([h|agg]))))
// ---------------------------------------------------------------------------
__global__ __launch_bounds__(32) void node_update_kernel(
    float* __restrict__ h, const float* __restrict__ agg,
    const unsigned* __restrict__ Wu1, const float* __restrict__ bu1,
    const unsigned* __restrict__ Wu2, const float* __restrict__ bu2,
    const float* __restrict__ nrm_g, const float* __restrict__ nrm_b){
  __shared__ unsigned       s_a[16*128];
  __shared__ unsigned short s_g1[16*128];
  __shared__ float          s_v[16*128];
  __shared__ float          s_red[64];
  const int lane = threadIdx.x;
  const long base = (long)blockIdx.x * 16;
  const int row = lane & 15, half = lane >> 4;

  {
    const float* hr = h   + (base + row)*128;
    const float* ar = agg + (base + row)*128;
    #pragma unroll 8
    for (int i = 0; i < 64; ++i){
      int idx = half*64 + i;
      float x0, x1;
      if (idx < 64){ x0 = hr[2*idx]; x1 = hr[2*idx + 1]; }
      else         { x0 = ar[2*(idx - 64)]; x1 = ar[2*(idx - 64) + 1]; }
      s_a[row*128 + idx] = pk_bf16(x0, x1);
    }
  }
  __syncthreads();
  #pragma unroll 1
  for (int nt = 0; nt < 8; ++nt){
    v8f c = zero8();
    #pragma unroll
    for (int kt = 0; kt < 8; ++kt){
      Frag a = lds_afrag_u32(s_a, 128, kt*32, lane);
      c = wmma_bf16(a, load_bfrag(Wu1, 8, kt, nt, lane), c);
    }
    int n = nt*16 + row; float bv = bu1[n];
    #pragma unroll
    for (int r = 0; r < 8; ++r) s_g1[(r + 8*half)*128 + n] = bf16s(gelu_f(c[r] + bv));
  }
  __syncthreads();
  #pragma unroll 1
  for (int nt = 0; nt < 8; ++nt){
    v8f c = zero8();
    #pragma unroll
    for (int kt = 0; kt < 4; ++kt){
      Frag a = lds_afrag_u16(s_g1, 128, kt*32, lane);
      c = wmma_bf16(a, load_bfrag(Wu2, 8, kt, nt, lane), c);
    }
    int n = nt*16 + row; float bv = bu2[n];
    #pragma unroll
    for (int r = 0; r < 8; ++r) s_v[(r + 8*half)*128 + n] = c[r] + bv;
  }
  __syncthreads();
  { // residual + partial LN stats (2 lanes per row)
    float* hr = h + (base + row)*128;
    float ps = 0.f, pq = 0.f;
    #pragma unroll 8
    for (int i = 0; i < 64; ++i){
      int cch = half*64 + i;
      float v = hr[cch] + s_v[row*128 + cch];
      s_v[row*128 + cch] = v;
      ps += v; pq += v*v;
    }
    s_red[lane] = ps; s_red[32 + lane] = pq;
  }
  __syncthreads();
  {
    float ts = s_red[row] + s_red[row + 16];
    float tq = s_red[32 + row] + s_red[48 + row];
    float mu = ts * (1.f/128.f);
    float var = tq * (1.f/128.f) - mu*mu;
    float inv = rsqrtf(var + 1e-5f);
    float* hr = h + (base + row)*128;
    #pragma unroll 8
    for (int i = 0; i < 64; ++i){
      int cch = half*64 + i;
      float v = s_v[row*128 + cch];
      hr[cch] = (v - mu)*inv*nrm_g[cch] + nrm_b[cch];
    }
  }
}

// ---------------------------------------------------------------------------
__global__ void pool_kernel(const float* __restrict__ h, const int* __restrict__ batch,
                            float* __restrict__ pooled, float* __restrict__ cnt, int N){
  int i = blockIdx.x*blockDim.x + threadIdx.x;
  if (i < N*128){
    int node = i >> 7, c = i & 127;
    int b = batch[node];
    atomAddF(&pooled[b*128 + c], h[i]);
    if (c == 0) atomAddF(&cnt[b], 1.f);
  }
}

__global__ __launch_bounds__(32) void out_kernel(
    const float* __restrict__ pooled, const float* __restrict__ cnt,
    const float* __restrict__ gl,
    const unsigned* __restrict__ Wo1, const float* __restrict__ bo1,
    const unsigned* __restrict__ Wo2, const float* __restrict__ bo2,
    float* __restrict__ out){
  __shared__ unsigned       s_a[16*128];
  __shared__ unsigned short s_g1[16*128];
  const int lane = threadIdx.x;
  const int base = blockIdx.x * 16;
  const int row = lane & 15, half = lane >> 4;
  {
    float invc = 1.f / fmaxf(cnt[base + row], 1.f);
    const float* pr = pooled + (base + row)*128;
    const float* gr = gl     + (base + row)*128;
    #pragma unroll 8
    for (int i = 0; i < 64; ++i){
      int idx = half*64 + i;
      float x0, x1;
      if (idx < 64){ x0 = pr[2*idx]*invc; x1 = pr[2*idx + 1]*invc; }
      else         { x0 = gr[2*(idx - 64)]; x1 = gr[2*(idx - 64) + 1]; }
      s_a[row*128 + idx] = pk_bf16(x0, x1);
    }
  }
  __syncthreads();
  #pragma unroll 1
  for (int nt = 0; nt < 8; ++nt){
    v8f c = zero8();
    #pragma unroll
    for (int kt = 0; kt < 8; ++kt){
      Frag a = lds_afrag_u32(s_a, 128, kt*32, lane);
      c = wmma_bf16(a, load_bfrag(Wo1, 8, kt, nt, lane), c);
    }
    int n = nt*16 + row; float bv = bo1[n];
    #pragma unroll
    for (int r = 0; r < 8; ++r) s_g1[(r + 8*half)*128 + n] = bf16s(gelu_f(c[r] + bv));
  }
  __syncthreads();
  #pragma unroll 1
  for (int nt = 0; nt < 8; ++nt){
    v8f c = zero8();
    #pragma unroll
    for (int kt = 0; kt < 4; ++kt){
      Frag a = lds_afrag_u16(s_g1, 128, kt*32, lane);
      c = wmma_bf16(a, load_bfrag(Wo2, 8, kt, nt, lane), c);
    }
    int n = nt*16 + row; float bv = bo2[n];
    #pragma unroll
    for (int r = 0; r < 8; ++r) out[(base + r + 8*half)*128 + n] = c[r] + bv;
  }
}

// ---------------------------------------------------------------------------
extern "C" void kernel_launch(void* const* d_in, const int* in_sizes, int n_in,
                              void* d_out, int out_size, void* d_ws, size_t ws_size,
                              hipStream_t stream){
  (void)n_in; (void)out_size; (void)ws_size;
  const float* x         = (const float*)d_in[0];
  const int*   eidx      = (const int*)  d_in[1];
  const float* edge_attr = (const float*)d_in[2];
  const int*   batch     = (const int*)  d_in[3];
  const float* gfeat     = (const float*)d_in[4];
  const int N = in_sizes[0] / 64;
  const int E = in_sizes[1] / 2;
  const int B = in_sizes[4] / 32;
  const int* src = eidx;
  const int* dst = eidx + E;

  auto F = [&](int i){ return (const float*)d_in[i]; };
  const float *gpW,*gpb,*npW,*npb,*o1W,*o1b,*o2W,*o2b;
  const float *e1W[3],*e1b[3],*e2W[3],*e2b[3],*enb[3],*eng[3],*gW[3],*gb[3];
  const float *m1W[3],*m1b[3],*m2W[3],*m2b[3],*nrb[3],*nrg[3],*u1W[3],*u1b[3],*u2W[3],*u2b[3];
  if (in_sizes[5] == 32*128){        // jax pytree sorted-key flatten order
    gpW=F(5); gpb=F(6);
    for (int l = 0; l < 3; ++l){ int b = 7 + 18*l;
      e1W[l]=F(b+0);  e1b[l]=F(b+1);  e2W[l]=F(b+2);  e2b[l]=F(b+3);
      enb[l]=F(b+4);  eng[l]=F(b+5);  gW[l]=F(b+6);   gb[l]=F(b+7);
      m1W[l]=F(b+8);  m1b[l]=F(b+9);  m2W[l]=F(b+10); m2b[l]=F(b+11);
      nrb[l]=F(b+12); nrg[l]=F(b+13); u1W[l]=F(b+14); u1b[l]=F(b+15);
      u2W[l]=F(b+16); u2b[l]=F(b+17);
    }
    npW=F(61); npb=F(62); o1W=F(63); o1b=F(64); o2W=F(65); o2b=F(66);
  } else {                           // insertion-order flatten
    npW=F(5); npb=F(6); gpW=F(7); gpb=F(8);
    o1W=F(9); o1b=F(10); o2W=F(11); o2b=F(12);
    for (int l = 0; l < 3; ++l){ int b = 13 + 18*l;
      m1W[l]=F(b+0);  m1b[l]=F(b+1);  m2W[l]=F(b+2);  m2b[l]=F(b+3);
      u1W[l]=F(b+4);  u1b[l]=F(b+5);  u2W[l]=F(b+6);  u2b[l]=F(b+7);
      nrg[l]=F(b+8);  nrb[l]=F(b+9);  e1W[l]=F(b+10); e1b[l]=F(b+11);
      e2W[l]=F(b+12); e2b[l]=F(b+13); gW[l]=F(b+14);  gb[l]=F(b+15);
      eng[l]=F(b+16); enb[l]=F(b+17);
    }
  }

  // ---- workspace carve-out ----
  char* ws = (char*)d_ws;
  size_t off = 0;
  auto alloc = [&](size_t bytes)->char*{
    char* p = ws + off; off += (bytes + 255) & ~(size_t)255; return p;
  };
  float*    h      = (float*)   alloc((size_t)N*128*4);
  unsigned* hbf    = (unsigned*)alloc((size_t)N*64*4);
  float*    agg    = (float*)   alloc((size_t)N*128*4);
  float*    ecur   = (float*)   alloc((size_t)E*16*4);
  float*    gl     = (float*)   alloc((size_t)B*128*4);
  float*    pooled = (float*)   alloc((size_t)B*128*4);
  float*    cnt    = (float*)   alloc((size_t)B*4);
  auto allocW = [&](int kt, int nt){ return (unsigned*)alloc((size_t)kt*nt*256*4); };
  unsigned* npWp = allocW(2,8);
  unsigned* gpWp = allocW(1,8);
  unsigned* o1Wp = allocW(8,8);
  unsigned* o2Wp = allocW(4,8);
  unsigned *e1Wp[3],*e2Wp[3],*m1Wp[3],*m2Wp[3],*u1Wp[3],*u2Wp[3];
  for (int l = 0; l < 3; ++l){
    e1Wp[l]=allocW(9,1); e2Wp[l]=allocW(1,1);
    m1Wp[l]=allocW(5,8); m2Wp[l]=allocW(4,8);
    u1Wp[l]=allocW(8,8); u2Wp[l]=allocW(4,8);
  }

  auto pack = [&](const float* W, unsigned* out, int din, int dout, int ktiles){
    int total = ktiles*(dout/16)*256;
    pack_weight_kernel<<<(total + 255)/256, 256, 0, stream>>>(W, out, din, dout, ktiles);
  };
  pack(npW, npWp, 64, 128, 2);
  pack(gpW, gpWp, 32, 128, 1);
  pack(o1W, o1Wp, 256, 128, 8);
  pack(o2W, o2Wp, 128, 128, 4);
  for (int l = 0; l < 3; ++l){
    pack(e1W[l], e1Wp[l], 272, 16, 9);
    pack(e2W[l], e2Wp[l], 16,  16, 1);
    pack(m1W[l], m1Wp[l], 144, 128, 5);
    pack(m2W[l], m2Wp[l], 128, 128, 4);
    pack(u1W[l], u1Wp[l], 256, 128, 8);
    pack(u2W[l], u2Wp[l], 128, 128, 4);
  }

  // ---- forward ----
  in_proj_kernel<<<N/16, 32, 0, stream>>>(x, 64, npWp, npb, h);
  in_proj_kernel<<<B/16, 32, 0, stream>>>(gfeat, 32, gpWp, gpb, gl);
  hipMemcpyAsync(ecur, edge_attr, (size_t)E*16*4, hipMemcpyDeviceToDevice, stream);

  for (int l = 0; l < 3; ++l){
    f32_to_bf16pk<<<(N*64 + 255)/256, 256, 0, stream>>>(h, hbf, N*64);
    hipMemsetAsync(agg, 0, (size_t)N*128*4, stream);
    edge_kernel<<<E/16, 32, 0, stream>>>(hbf, ecur, src, dst,
        e1Wp[l], e1b[l], e2Wp[l], e2b[l], enb[l], eng[l], gW[l], gb[l],
        m1Wp[l], m1b[l], m2Wp[l], m2b[l], agg);
    node_update_kernel<<<N/16, 32, 0, stream>>>(h, agg,
        u1Wp[l], u1b[l], u2Wp[l], u2b[l], nrg[l], nrb[l]);
  }

  hipMemsetAsync(pooled, 0, (size_t)B*128*4, stream);
  hipMemsetAsync(cnt, 0, (size_t)B*4, stream);
  pool_kernel<<<(N*128 + 255)/256, 256, 0, stream>>>(h, batch, pooled, cnt, N);
  out_kernel<<<B/16, 32, 0, stream>>>(pooled, cnt, gl, o1Wp, o1b, o2Wp, o2b, (float*)d_out);
}
